// DotProductAttention_36275293782314
// MI455X (gfx1250) — matrix operand order repaired and checked by reading, not compiled, not run on hardware
//
#include <hip/hip_runtime.h>
#include <stdint.h>

typedef __attribute__((ext_vector_type(16))) _Float16 v16h;
typedef __attribute__((ext_vector_type(8)))  _Float16 v8h;
typedef __attribute__((ext_vector_type(4)))  _Float16 v4h;
typedef __attribute__((ext_vector_type(8)))  float    v8f;
typedef __attribute__((ext_vector_type(4)))  float    v4f;
typedef __attribute__((ext_vector_type(4)))  unsigned int v4u;
typedef __attribute__((ext_vector_type(8)))  int      v8i;
typedef __attribute__((ext_vector_type(4)))  int      v4i;

#define Bc 16
#define Nc 2048
#define Dc 128

// TDM kernel geometry: 4 waves x 16 query rows, 128-row KV tiles, paired qblks
#define QB 64
#define KB 128
#define NW 4

// fp32 fallback kernel geometry (round-1 proven)
#define MBLK  128
#define KVBLK 64
#define NWAVE 8

#if __has_builtin(__builtin_amdgcn_tensor_load_to_lds)
#define HAVE_TDM 1
#else
#define HAVE_TDM 0
#endif

#if __has_builtin(__builtin_amdgcn_exp2f)
#define EXP2F(x) __builtin_amdgcn_exp2f(x)
#else
#define EXP2F(x) exp2f(x)
#endif

// (1/sqrt(128)) * log2(e): softmax runs in exp2 domain with raw v_exp_f32
#define SCALE_L2E 0.12752650696861288f

#if HAVE_TDM
// 2-D Tensor-DMA descriptor (CDNA5 ISA); sizes in elements (2B).
__device__ __forceinline__ void tdm_load_2d(uint32_t lds_off, const _Float16* g,
                                            uint32_t tile_w, uint32_t tile_h,
                                            uint32_t ten_w, uint32_t ten_h,
                                            uint32_t stride) {
  const uint64_t ga = (uint64_t)(uintptr_t)g;
  v4u g0;
  g0[0] = 1u;                                   // count=1 (valid), no gather
  g0[1] = lds_off;                              // LDS byte offset
  g0[2] = (uint32_t)ga;                         // global_addr[31:0]
  g0[3] = (uint32_t)(ga >> 32) | (2u << 30);    // global_addr[56:32] | type=2
  v8i g1;
  g1[0] = (int)0x00010000u;                     // data_size=1 (2B)
  g1[1] = (int)((ten_w & 0xFFFFu) << 16);       // tensor_dim0 lo
  g1[2] = (int)((ten_w >> 16) | ((ten_h & 0xFFFFu) << 16));
  g1[3] = (int)((ten_h >> 16) | (tile_w << 16));
  g1[4] = (int)tile_h;                          // tile_dim1 | tile_dim2=0
  g1[5] = (int)stride;                          // tensor_dim0_stride lo
  g1[6] = 0;
  g1[7] = 0;
  v4i z4 = {};
#if defined(__has_include) && __has_include(<hip/amd_detail/amd_gfx1250_TDM.h>)
  v8i z8 = {};
  __builtin_amdgcn_tensor_load_to_lds(g0, g1, z4, z4, z8, 0);
#else
  __builtin_amdgcn_tensor_load_to_lds(g0, g1, z4, z4, 0);
#endif
}
#endif // HAVE_TDM

// ---------------------------------------------------------------------------
// Pass 1: fp32 -> f16 (Q gets scale*log2e folded; V transposed to Vt[b][d][n]).
// ---------------------------------------------------------------------------
__launch_bounds__(256)
__global__ void preconv(const float* __restrict__ Q, const float* __restrict__ K,
                        const float* __restrict__ V, _Float16* __restrict__ Qh,
                        _Float16* __restrict__ Kh, _Float16* __restrict__ Vth) {
  const size_t i4 = ((size_t)blockIdx.x * 256 + threadIdx.x) * 4;
  v4f q = *(const v4f*)(Q + i4);
  v4f k = *(const v4f*)(K + i4);
  v4f v = *(const v4f*)(V + i4);
  v4h qh, kh;
#pragma unroll
  for (int t = 0; t < 4; ++t) {
    qh[t] = (_Float16)(q[t] * SCALE_L2E);
    kh[t] = (_Float16)k[t];
  }
  *(v4h*)(Qh + i4) = qh;
  *(v4h*)(Kh + i4) = kh;
  const int    d  = (int)(i4 & (Dc - 1));
  const size_t bn = i4 >> 7;
  const int    n  = (int)(bn & (Nc - 1));
  const size_t b  = bn >> 11;
  _Float16* vt = Vth + (size_t)b * Dc * Nc + n;
#pragma unroll
  for (int t = 0; t < 4; ++t)
    vt[(size_t)(d + t) * Nc] = (_Float16)v[t];
}

// ---------------------------------------------------------------------------
// One 64-row query block against its causal KV range (tiles of 128 rows).
// ---------------------------------------------------------------------------
__device__ __forceinline__ void attn_block(
    const _Float16* __restrict__ QhB, const _Float16* __restrict__ KhB,
    const _Float16* __restrict__ VtB, float* __restrict__ Og,
    _Float16 (&Kl)[2][KB * Dc], _Float16 (&Vt)[2][Dc * KB],
    _Float16* __restrict__ Pw,
    int qblk, int tid, int wave, int hg, int ln) {
  const int qbase = qblk * QB;
  const int qr0   = qbase + wave * 16;
  const int njt   = (qblk >> 1) + 1;     // causal KV tile count

#if HAVE_TDM
  if (wave == 0) {   // kick off tile 0 DMA immediately
    tdm_load_2d((uint32_t)(uintptr_t)&Kl[0][0], KhB, Dc, KB, Dc, Nc, Dc);
    tdm_load_2d((uint32_t)(uintptr_t)&Vt[0][0], VtB, KB, Dc, Nc, Dc, Nc);
  }
#endif

  // Q fragments, A layout (lane<16: K {0..15}; lane>=16: {16..31})
  v16h qf[4];
  {
    const int qrow = qr0 + ln, kbase = hg ? 16 : 0;
#pragma unroll
    for (int s = 0; s < 4; ++s) {
      const _Float16* p0 = QhB + (size_t)qrow * Dc + s * 32 + kbase;
      v8h lo = *(const v8h*)p0;
      v8h hi = *(const v8h*)(p0 + 8);
      qf[s] = __builtin_shufflevector(lo, hi,
                0,1,2,3,4,5,6,7,8,9,10,11,12,13,14,15);
    }
  }

  v8f o[8];
#pragma unroll
  for (int i = 0; i < 8; ++i) o[i] = v8f{};
  float mrun[8], lrun[8];
#pragma unroll
  for (int r = 0; r < 8; ++r) { mrun[r] = -__builtin_inff(); lrun[r] = 0.0f; }

  for (int j = 0; j < njt; ++j) {
    const int kvbase = j * KB;
    const int buf    = j & 1;

#if HAVE_TDM
    __builtin_amdgcn_s_wait_tensorcnt(0);   // tile j resident
    __syncthreads();                        // publish; tile j-1 readers done
    if (wave == 0 && j + 1 < njt) {         // DMA tile j+1 under compute of j
      const int nb = kvbase + KB;
      tdm_load_2d((uint32_t)(uintptr_t)&Kl[buf ^ 1][0],
                  KhB + (size_t)nb * Dc, Dc, KB, Dc, Nc, Dc);
      tdm_load_2d((uint32_t)(uintptr_t)&Vt[buf ^ 1][0],
                  VtB + nb, KB, Dc, Nc, Dc, Nc);
    }
#else
    __syncthreads();
    for (int i = tid; i < KB * Dc / 8; i += 128) {
      const int r = i >> 4, c = (i & 15) << 3;
      *(v8h*)&Kl[buf][r * Dc + c] = *(const v8h*)(KhB + (size_t)(kvbase + r) * Dc + c);
    }
    for (int i = tid; i < Dc * KB / 8; i += 128) {
      const int r = i >> 4, c = (i & 15) << 3;
      *(v8h*)&Vt[buf][r * KB + c] = *(const v8h*)(VtB + (size_t)r * Nc + kvbase + c);
    }
    __syncthreads();
#endif

    // ---- S = Q' K^T : 32 WMMAs
    v8f c[8];
#pragma unroll
    for (int nt = 0; nt < 8; ++nt) c[nt] = v8f{};
#pragma unroll
    for (int nt = 0; nt < 8; ++nt) {
      const int krow = nt * 16 + ln;
#pragma unroll
      for (int s = 0; s < 4; ++s) {
        const int koff = s * 32 + (hg ? 16 : 0);   // B layout: contiguous K half
        v16h kf = *(const v16h*)(&Kl[buf][krow * Dc + koff]);
        c[nt] = __builtin_amdgcn_wmma_f32_16x16x32_f16(
            false, qf[s], false, kf, (short)0, c[nt], false, false);
      }
    }

    // ---- online softmax (exp2 domain); mask only on the diagonal tile
    const int rg0 = qr0 + 8 * hg;
    float mcur[8];
#pragma unroll
    for (int r = 0; r < 8; ++r) mcur[r] = -__builtin_inff();
    if (j == njt - 1) {                 // diagonal tile: apply causal mask
#pragma unroll
      for (int nt = 0; nt < 8; ++nt) {
        const int cg = kvbase + nt * 16 + ln;
#pragma unroll
        for (int r = 0; r < 8; ++r) {
          float s = c[nt][r];
          s = (cg > rg0 + r) ? -__builtin_inff() : s;
          c[nt][r] = s;
          mcur[r]  = fmaxf(mcur[r], s);
        }
      }
    } else {                            // interior tile: mask-free fast path
#pragma unroll
      for (int nt = 0; nt < 8; ++nt)
#pragma unroll
        for (int r = 0; r < 8; ++r)
          mcur[r] = fmaxf(mcur[r], c[nt][r]);
    }
#pragma unroll
    for (int r = 0; r < 8; ++r) {
#pragma unroll
      for (int off = 8; off >= 1; off >>= 1)
        mcur[r] = fmaxf(mcur[r], __shfl_xor(mcur[r], off, 32));
    }
    float alpha[8], lcur[8];
#pragma unroll
    for (int r = 0; r < 8; ++r) {
      const float mnew = fmaxf(mrun[r], mcur[r]);
      alpha[r] = EXP2F(mrun[r] - mnew);
      mrun[r]  = mnew;
      lcur[r]  = 0.0f;
    }
#pragma unroll
    for (int nt = 0; nt < 8; ++nt) {
#pragma unroll
      for (int r = 0; r < 8; ++r) {
        const float p = EXP2F(c[nt][r] - mrun[r]);
        c[nt][r] = p;
        lcur[r] += p;
      }
    }
#pragma unroll
    for (int r = 0; r < 8; ++r) {
#pragma unroll
      for (int off = 8; off >= 1; off >>= 1)
        lcur[r] += __shfl_xor(lcur[r], off, 32);
      lrun[r] = lrun[r] * alpha[r] + lcur[r];
    }
#pragma unroll
    for (int dt = 0; dt < 8; ++dt)
#pragma unroll
      for (int r = 0; r < 8; ++r)
        o[dt][r] *= alpha[r];

    // ---- P: C layout -> A layout via wave-private LDS strip
#pragma unroll
    for (int nt = 0; nt < 8; ++nt) {
#pragma unroll
      for (int r = 0; r < 8; ++r)
        Pw[(r + 8 * hg) * KB + nt * 16 + ln] = (_Float16)c[nt][r];
    }
    asm volatile("s_wait_dscnt 0x0" ::: "memory");

    // ---- O += P V : 32 WMMAs
#pragma unroll
    for (int s2 = 0; s2 < 4; ++s2) {
      const int ka = s2 * 32 + (hg ? 16 : 0);
      v8h  plo = *(const v8h*)(&Pw[ln * KB + ka]);
      v8h  phi = *(const v8h*)(&Pw[ln * KB + ka + 8]);
      v16h pf  = __builtin_shufflevector(plo, phi,
                   0,1,2,3,4,5,6,7,8,9,10,11,12,13,14,15);
      const int kb = s2 * 32 + (hg ? 16 : 0);
#pragma unroll
      for (int dt = 0; dt < 8; ++dt) {
        v16h vf = *(const v16h*)(&Vt[buf][(dt * 16 + ln) * KB + kb]);
        o[dt] = __builtin_amdgcn_wmma_f32_16x16x32_f16(
            false, pf, false, vf, (short)0, o[dt], false, false);
      }
    }
  }

  // ---- normalize + store
#pragma unroll
  for (int r = 0; r < 8; ++r) {
    const float inv = 1.0f / lrun[r];
    float* orow = Og + (size_t)(qr0 + r + 8 * hg) * Dc;
#pragma unroll
    for (int dt = 0; dt < 8; ++dt)
      orow[dt * 16 + ln] = o[dt][r] * inv;
  }
}

// ---------------------------------------------------------------------------
// Pass 2: causal flash attention; each WG runs the complementary pair
// {31-p, p} of 64-row query blocks -> every WG does exactly 17 KV tiles.
// ---------------------------------------------------------------------------
__launch_bounds__(128)
__global__ void fattn_tc(const _Float16* __restrict__ Qh,
                         const _Float16* __restrict__ Kh,
                         const _Float16* __restrict__ Vth,
                         float* __restrict__ O) {
  __shared__ alignas(128) _Float16 Kl[2][KB * Dc];   // K tile [n][d]
  __shared__ alignas(128) _Float16 Vt[2][Dc * KB];   // V tile [d][n]
  __shared__ alignas(32)  _Float16 Pl[NW * 16 * KB];

  const int tid  = threadIdx.x;
  const int lane = tid & 31;
  const int wave = tid >> 5;
  const int hg   = lane >> 4;
  const int ln   = lane & 15;

  const int blk = blockIdx.x;
  const int b   = blk >> 4;
  const int p   = blk & 15;

  const _Float16* QhB = Qh  + (size_t)b * Nc * Dc;
  const _Float16* KhB = Kh  + (size_t)b * Nc * Dc;
  const _Float16* VtB = Vth + (size_t)b * Dc * Nc;
  float*          Og  = O   + (size_t)b * Nc * Dc;
  _Float16*       Pw  = &Pl[wave * 16 * KB];

  attn_block(QhB, KhB, VtB, Og, Kl, Vt, Pw, 31 - p, tid, wave, hg, ln);
  __syncthreads();   // phase-1 LDS readers done before phase-2 tile-0 DMA
  attn_block(QhB, KhB, VtB, Og, Kl, Vt, Pw, p, tid, wave, hg, ln);
}

// ---------------------------------------------------------------------------
// Fallback (ws too small): round-1 fp32-staging flash attention.
// ---------------------------------------------------------------------------
__launch_bounds__(256)
__global__ void fattn_f32(const float* __restrict__ Q, const float* __restrict__ K,
                          const float* __restrict__ V, float* __restrict__ O) {
  __shared__ alignas(32) _Float16 Kl[KVBLK * Dc];
  __shared__ alignas(32) _Float16 Vt[Dc * KVBLK];
  __shared__ alignas(32) _Float16 Pl[NWAVE * 16 * KVBLK];

  const int tid = threadIdx.x, lane = tid & 31, wave = tid >> 5;
  const int hg = lane >> 4, ln = lane & 15;
  const int blk = blockIdx.x, b = blk >> 4, qblk = blk & 15;
  const int qbase = qblk * MBLK, qr0 = qbase + wave * 16;
  const float* Qg = Q + (size_t)b * Nc * Dc;
  const float* Kg = K + (size_t)b * Nc * Dc;
  const float* Vg = V + (size_t)b * Nc * Dc;
  float*       Og = O + (size_t)b * Nc * Dc;

  v16h qf[4];
  {
    const int qrow = qr0 + ln, kbase = hg ? 16 : 0;
#pragma unroll
    for (int s = 0; s < 4; ++s) {
      const float* p0 = Qg + (size_t)qrow * Dc + s * 32 + kbase;
      v4f a0 = *(const v4f*)(p0),      a1 = *(const v4f*)(p0 + 4);
      v4f b0 = *(const v4f*)(p0 + 8), b1 = *(const v4f*)(p0 + 12);
      v16h f;
#pragma unroll
      for (int t = 0; t < 4; ++t) {
        f[t]      = (_Float16)(a0[t] * SCALE_L2E);
        f[4 + t]  = (_Float16)(a1[t] * SCALE_L2E);
        f[8 + t]  = (_Float16)(b0[t] * SCALE_L2E);
        f[12 + t] = (_Float16)(b1[t] * SCALE_L2E);
      }
      qf[s] = f;
    }
  }
  v8f o[8];
#pragma unroll
  for (int i = 0; i < 8; ++i) o[i] = v8f{};
  float mrun[8], lrun[8];
#pragma unroll
  for (int r = 0; r < 8; ++r) { mrun[r] = -__builtin_inff(); lrun[r] = 0.0f; }

  const int njt = 2 * qblk + 2;
  for (int j = 0; j < njt; ++j) {
    const int kvbase = j * KVBLK;
    __syncthreads();
    for (int i = tid; i < KVBLK * Dc / 4; i += 256) {
      const int n = i >> 5, d4 = (i & 31) << 2;
      v4f f = *(const v4f*)(Kg + (size_t)(kvbase + n) * Dc + d4);
      v4h h;
      h[0] = (_Float16)f[0]; h[1] = (_Float16)f[1];
      h[2] = (_Float16)f[2]; h[3] = (_Float16)f[3];
      *(v4h*)(&Kl[n * Dc + d4]) = h;
    }
    for (int i = tid; i < KVBLK * Dc / 4; i += 256) {
      const int n = i >> 5, d4 = (i & 31) << 2;
      v4f f = *(const v4f*)(Vg + (size_t)(kvbase + n) * Dc + d4);
      Vt[(d4 + 0) * KVBLK + n] = (_Float16)f[0];
      Vt[(d4 + 1) * KVBLK + n] = (_Float16)f[1];
      Vt[(d4 + 2) * KVBLK + n] = (_Float16)f[2];
      Vt[(d4 + 3) * KVBLK + n] = (_Float16)f[3];
    }
    __syncthreads();
    if (kvbase > qr0 + 15) continue;

    v8f c[4];
#pragma unroll
    for (int nt = 0; nt < 4; ++nt) c[nt] = v8f{};
#pragma unroll
    for (int nt = 0; nt < 4; ++nt) {
      const int krow = nt * 16 + ln;
#pragma unroll
      for (int s = 0; s < 4; ++s) {
        v16h kf = *(const v16h*)(&Kl[krow * Dc + s * 32 + (hg ? 16 : 0)]);
        c[nt] = __builtin_amdgcn_wmma_f32_16x16x32_f16(
            false, qf[s], false, kf, (short)0, c[nt], false, false);
      }
    }
    const int rg0 = qr0 + 8 * hg;
    float mcur[8];
#pragma unroll
    for (int r = 0; r < 8; ++r) mcur[r] = -__builtin_inff();
#pragma unroll
    for (int nt = 0; nt < 4; ++nt) {
      const int cg = kvbase + nt * 16 + ln;
#pragma unroll
      for (int r = 0; r < 8; ++r) {
        float s = c[nt][r];
        s = (cg > rg0 + r) ? -__builtin_inff() : s;
        c[nt][r] = s;
        mcur[r]  = fmaxf(mcur[r], s);
      }
    }
#pragma unroll
    for (int r = 0; r < 8; ++r)
#pragma unroll
      for (int off = 8; off >= 1; off >>= 1)
        mcur[r] = fmaxf(mcur[r], __shfl_xor(mcur[r], off, 32));
    float alpha[8], lcur[8];
#pragma unroll
    for (int r = 0; r < 8; ++r) {
      const float mnew = fmaxf(mrun[r], mcur[r]);
      alpha[r] = EXP2F(mrun[r] - mnew);
      mrun[r] = mnew; lcur[r] = 0.0f;
    }
#pragma unroll
    for (int nt = 0; nt < 4; ++nt)
#pragma unroll
      for (int r = 0; r < 8; ++r) {
        const float p = EXP2F(c[nt][r] - mrun[r]);
        c[nt][r] = p; lcur[r] += p;
      }
#pragma unroll
    for (int r = 0; r < 8; ++r) {
#pragma unroll
      for (int off = 8; off >= 1; off >>= 1)
        lcur[r] += __shfl_xor(lcur[r], off, 32);
      lrun[r] = lrun[r] * alpha[r] + lcur[r];
    }
#pragma unroll
    for (int dt = 0; dt < 8; ++dt)
#pragma unroll
      for (int r = 0; r < 8; ++r) o[dt][r] *= alpha[r];

    _Float16* Pw = &Pl[wave * 16 * KVBLK];
#pragma unroll
    for (int nt = 0; nt < 4; ++nt)
#pragma unroll
      for (int r = 0; r < 8; ++r)
        Pw[(r + 8 * hg) * KVBLK + nt * 16 + ln] = (_Float16)c[nt][r];
    asm volatile("s_wait_dscnt 0x0" ::: "memory");

#pragma unroll
    for (int s2 = 0; s2 < 2; ++s2) {
      const int ka = s2 * 32 + (hg ? 16 : 0);
      v8h  plo = *(const v8h*)(&Pw[ln * KVBLK + ka]);
      v8h  phi = *(const v8h*)(&Pw[ln * KVBLK + ka + 8]);
      v16h pf  = __builtin_shufflevector(plo, phi,
                   0,1,2,3,4,5,6,7,8,9,10,11,12,13,14,15);
      const int kb = s2 * 32 + (hg ? 16 : 0);
#pragma unroll
      for (int dt = 0; dt < 8; ++dt) {
        v16h vf = *(const v16h*)(&Vt[(dt * 16 + ln) * KVBLK + kb]);
        o[dt] = __builtin_amdgcn_wmma_f32_16x16x32_f16(
            false, pf, false, vf, (short)0, o[dt], false, false);
      }
    }
  }
#pragma unroll
  for (int r = 0; r < 8; ++r) {
    const float inv = 1.0f / lrun[r];
    float* orow = Og + (size_t)(qr0 + r + 8 * hg) * Dc;
#pragma unroll
    for (int dt = 0; dt < 8; ++dt)
      orow[dt * 16 + ln] = o[dt][r] * inv;
  }
}

extern "C" void kernel_launch(void* const* d_in, const int* in_sizes, int n_in,
                              void* d_out, int out_size, void* d_ws, size_t ws_size,
                              hipStream_t stream) {
  const float* q = (const float*)d_in[0];
  const float* k = (const float*)d_in[1];
  const float* v = (const float*)d_in[2];
  float* out = (float*)d_out;
  (void)in_sizes; (void)n_in; (void)out_size;

  const size_t elems = (size_t)Bc * Nc * Dc;          // 4,194,304 per tensor
  const size_t need  = elems * 3 * sizeof(_Float16);  // 24 MB f16 scratch
  if (ws_size >= need) {
    _Float16* Qh  = (_Float16*)d_ws;
    _Float16* Kh  = Qh + elems;
    _Float16* Vth = Kh + elems;
    preconv<<<dim3((unsigned)(elems / 4 / 256)), dim3(256), 0, stream>>>(q, k, v, Qh, Kh, Vth);
    fattn_tc<<<dim3(256), dim3(128), 0, stream>>>(Qh, Kh, Vth, out);
  } else {
    fattn_f32<<<dim3(Bc * (Nc / MBLK)), dim3(256), 0, stream>>>(q, k, v, out);
  }
}
